// CausalSelfAttention_30674656428129
// MI455X (gfx1250) — matrix operand order repaired and checked by reading, not compiled, not run on hardware
//
#include <hip/hip_runtime.h>
#include <stdint.h>

typedef _Float16 h16;
typedef h16   v16h __attribute__((ext_vector_type(16)));
typedef h16   v8h  __attribute__((ext_vector_type(8)));
typedef float v8f  __attribute__((ext_vector_type(8)));
typedef int   v4i  __attribute__((ext_vector_type(4)));

#if defined(__gfx1250__) && __has_builtin(__builtin_amdgcn_global_load_async_to_lds_b128)
#define HAVE_ASYNC_LDS 1
#else
#define HAVE_ASYNC_LDS 0
#endif

static __device__ __forceinline__ v16h cat8(v8h lo, v8h hi) {
  return __builtin_shufflevector(lo, hi, 0,1,2,3,4,5,6,7,8,9,10,11,12,13,14,15);
}
static __device__ __forceinline__ v8f vzero8() {
  v8f z;
#pragma unroll
  for (int i = 0; i < 8; ++i) z[i] = 0.0f;
  return z;
}

#if HAVE_ASYNC_LDS
typedef __attribute__((address_space(1))) v4i* gptr4;
typedef __attribute__((address_space(3))) v4i* lptr4;
static __device__ __forceinline__ void async_copy16(h16* lds, const h16* g) {
  __builtin_amdgcn_global_load_async_to_lds_b128((gptr4)g, (lptr4)lds, 0, 0);
}
static __device__ __forceinline__ void wait_async0() {
#if __has_builtin(__builtin_amdgcn_s_wait_asynccnt)
  __builtin_amdgcn_s_wait_asynccnt(0);
#else
  asm volatile("s_wait_asynccnt 0" ::: "memory");
#endif
}
#endif

// 16-lane (DPP row) XOR-tree reductions: xor1, xor2, xor7, xor15 stay in-row.
template <int CTRL>
static __device__ __forceinline__ float dppf(float x) {
  return __builtin_bit_cast(float,
      __builtin_amdgcn_update_dpp(0, __builtin_bit_cast(int, x), CTRL, 0xF, 0xF, true));
}
static __device__ __forceinline__ float rowmax16(float x) {
  x = fmaxf(x, dppf<0xB1>(x));   // quad_perm(1,0,3,2)
  x = fmaxf(x, dppf<0x4E>(x));   // quad_perm(2,3,0,1)
  x = fmaxf(x, dppf<0x141>(x));  // row_half_mirror
  x = fmaxf(x, dppf<0x140>(x));  // row_mirror
  return x;
}
static __device__ __forceinline__ float rowsum16(float x) {
  x += dppf<0xB1>(x);
  x += dppf<0x4E>(x);
  x += dppf<0x141>(x);
  x += dppf<0x140>(x);
  return x;
}

// ---------------------------------------------------------------------------
// f32 -> f16 convert
// ---------------------------------------------------------------------------
__global__ void cvt_f32_f16(const float* __restrict__ src, h16* __restrict__ dst, int n) {
  int i = (blockIdx.x * blockDim.x + threadIdx.x) * 4;
  if (i < n) {
    float4 f = *(const float4*)(src + i);
    dst[i + 0] = (h16)f.x;
    dst[i + 1] = (h16)f.y;
    dst[i + 2] = (h16)f.z;
    dst[i + 3] = (h16)f.w;
  }
}

// ---------------------------------------------------------------------------
// WMMA GEMM: C[M,N] = A[M,K] * Bt[N,K]^T   (f16 in, f32 accum)
// Block tile 128x128x32, 256 threads = 8 waves, wave tile 64x32 (4x2 WMMA).
// ---------------------------------------------------------------------------
template <bool SPLIT_QKV>
__global__ __launch_bounds__(256) void gemm_f16(
    const h16* __restrict__ A, const h16* __restrict__ Bt, float* __restrict__ Cf,
    h16* __restrict__ Qo, h16* __restrict__ Ko, h16* __restrict__ Vo,
    int M, int N, int Kdim) {
  constexpr int BM = 128, BN = 128, BK = 32, PITCH = 40;  // 80B pitch, 16B aligned
  __shared__ alignas(16) h16 As[BM * PITCH];
  __shared__ alignas(16) h16 Bs[BN * PITCH];
  (void)M;

  const int tid  = threadIdx.x;
  const int lane = tid & 31;
  const int wave = tid >> 5;
  const int wm = (wave >> 2) * 64;
  const int wn = (wave & 3) * 32;
  const int m0 = blockIdx.y * BM;
  const int n0 = blockIdx.x * BN;
  const int ln15 = lane & 15;
  const int kb8  = (lane >> 4) * 8;   // A-frag K base (ISA 16-bit A 16x32 layout)
  const int kb16 = (lane >> 4) * 16;  // unused; B-frag also uses kb8

  v8f acc[4][2];
#pragma unroll
  for (int i = 0; i < 4; ++i)
#pragma unroll
    for (int j = 0; j < 2; ++j) acc[i][j] = vzero8();

  const int ldr = tid >> 2;          // staging row (pass 0)
  const int ldc = (tid & 3) * 8;     // staging col

  for (int k0 = 0; k0 < Kdim; k0 += BK) {
#pragma unroll
    for (int it = 0; it < 2; ++it) {
      int idx = tid + it * 256;
      int r = idx >> 2;
      int c = (idx & 3) * 8;
      const h16* ga = &A [(size_t)(m0 + r) * Kdim + k0 + c];
      const h16* gb = &Bt[(size_t)(n0 + r) * Kdim + k0 + c];
#if HAVE_ASYNC_LDS
      async_copy16(&As[r * PITCH + c], ga);
      async_copy16(&Bs[r * PITCH + c], gb);
#else
      *(v8h*)&As[r * PITCH + c] = *(const v8h*)ga;
      *(v8h*)&Bs[r * PITCH + c] = *(const v8h*)gb;
#endif
    }
    if (k0 + BK < Kdim) {  // L2 prefetch of next K-tile
      __builtin_prefetch(&A [(size_t)(m0 + ldr) * Kdim + k0 + BK + ldc], 0, 1);
      __builtin_prefetch(&Bt[(size_t)(n0 + ldr) * Kdim + k0 + BK + ldc], 0, 1);
    }
#if HAVE_ASYNC_LDS
    wait_async0();
#endif
    __syncthreads();

    v16h af[4], bfr[2];
#pragma unroll
    for (int mi = 0; mi < 4; ++mi) {
      int row = wm + mi * 16 + ln15;
      af[mi] = cat8(*(const v8h*)&As[row * PITCH + kb8],
                    *(const v8h*)&As[row * PITCH + kb8 + 16]);
    }
#pragma unroll
    for (int ni = 0; ni < 2; ++ni) {
      int row = wn + ni * 16 + ln15;
      bfr[ni] = cat8(*(const v8h*)&Bs[row * PITCH + kb8],
                     *(const v8h*)&Bs[row * PITCH + kb8 + 16]);
    }
#pragma unroll
    for (int mi = 0; mi < 4; ++mi)
#pragma unroll
      for (int ni = 0; ni < 2; ++ni)
        acc[mi][ni] = __builtin_amdgcn_wmma_f32_16x16x32_f16(
            false, af[mi], false, bfr[ni], (short)0, acc[mi][ni], false, false);
    __syncthreads();
  }

  const int rofs = (lane >> 4) * 8;
#pragma unroll
  for (int mi = 0; mi < 4; ++mi)
#pragma unroll
    for (int ni = 0; ni < 2; ++ni)
#pragma unroll
      for (int r = 0; r < 8; ++r) {
        int row = m0 + wm + mi * 16 + r + rofs;
        int col = n0 + wn + ni * 16 + ln15;
        float v = acc[mi][ni][r];
        if (SPLIT_QKV) {
          int sel = col >> 10, c = col & 1023;
          h16* dst = (sel == 0) ? Qo : (sel == 1 ? Ko : Vo);
          dst[(size_t)row * 1024 + c] = (h16)v;
        } else {
          Cf[(size_t)row * N + col] = v;
        }
      }
}

// ---------------------------------------------------------------------------
// Flash attention, causal. Grid: (T/64, B*H), 128 threads = 4 waves.
// 64-key blocks: softmax bookkeeping, barriers and DPP reductions amortized
// over 2x more keys per iteration. K/V tiles shared by all waves
// (block-uniform trip count); fully-masked sub-tiles contribute exactly zero.
// Softmax scale 1/8 is pre-folded into the Q fragments (exact in f16).
// ---------------------------------------------------------------------------
__global__ __launch_bounds__(128) void flash_attn(
    const h16* __restrict__ Qm, const h16* __restrict__ Km,
    const h16* __restrict__ Vm, h16* __restrict__ Om) {
  constexpr int T = 2048, C = 1024, D = 64;
  __shared__ alignas(16) h16 kbS[64 * 64];      // K tile  [key][d]
  __shared__ alignas(16) h16 vtS[64 * 64];      // V tile transposed [d][key]
  __shared__ alignas(16) h16 pbS[4][16 * 64];   // per-wave P

  const int tid  = threadIdx.x;
  const int lane = tid & 31;
  const int wave = tid >> 5;
  const int b = blockIdx.y >> 4;
  const int h = blockIdx.y & 15;
  const int qbase = blockIdx.x * 64 + wave * 16;
  h16* pb = pbS[wave];

  const int ln15 = lane & 15;
  const int hi   = lane >> 4;
  const int kb8  = hi * 8;
  const int kb16 = hi * 16;

  // Q A-fragments for d-chunks [0,32) and [32,64), pre-scaled by 1/sqrt(D)
  v16h aq[2];
  {
    const h16* qp = Qm + ((size_t)(b * T + qbase + ln15)) * C + h * D + kb8;
#pragma unroll
    for (int dc = 0; dc < 2; ++dc) {
      v16h q = cat8(*(const v8h*)(qp + dc * 32), *(const v8h*)(qp + dc * 32 + 16));
      aq[dc] = q * (h16)0.125f;
    }
  }

  float mrow[8], lrow[8];
  v8f oacc[4];
#pragma unroll
  for (int r = 0; r < 8; ++r) { mrow[r] = -3.0e38f; lrow[r] = 0.0f; }
#pragma unroll
  for (int dn = 0; dn < 4; ++dn) oacc[dn] = vzero8();

  const int tmax = blockIdx.x * 64 + 63;  // uniform across the block
  for (int t0 = 0; t0 <= tmax; t0 += 64) {
    __syncthreads();  // previous iteration's readers are done
    // Stage K tile [64][64]: 512 16B chunks, 4 per thread
#pragma unroll
    for (int it = 0; it < 4; ++it) {
      int idx = tid + it * 128;
      int r = idx >> 3, c = (idx & 7) * 8;
      const h16* g = &Km[((size_t)(b * T + t0 + r)) * C + h * D + c];
#if HAVE_ASYNC_LDS
      async_copy16(&kbS[r * 64 + c], g);
#else
      *(v8h*)&kbS[r * 64 + c] = *(const v8h*)g;
#endif
    }
    // Stage V tile transposed: vt[d][key]
#pragma unroll
    for (int it = 0; it < 4; ++it) {
      int idx = tid + it * 128;
      int r = idx >> 3, c = (idx & 7) * 8;
      v8h vv = *(const v8h*)&Vm[((size_t)(b * T + t0 + r)) * C + h * D + c];
#pragma unroll
      for (int e = 0; e < 8; ++e) vtS[(c + e) * 64 + r] = vv[e];
    }
#if HAVE_ASYNC_LDS
    wait_async0();
#endif
    __syncthreads();

    // S = Q K^T : four 16x16 key sub-tiles, K-dim 64 = 2 WMMAs each
    v8f s[4];
#pragma unroll
    for (int st = 0; st < 4; ++st) {
      int n = st * 16 + ln15;
      v16h b0 = cat8(*(const v8h*)&kbS[n * 64 + kb8],
                     *(const v8h*)&kbS[n * 64 + kb8 + 16]);
      v16h b1 = cat8(*(const v8h*)&kbS[n * 64 + 32 + kb8],
                     *(const v8h*)&kbS[n * 64 + 32 + kb8 + 16]);
      v8f z = vzero8();
      z = __builtin_amdgcn_wmma_f32_16x16x32_f16(false, aq[0], false, b0, (short)0, z, false, false);
      z = __builtin_amdgcn_wmma_f32_16x16x32_f16(false, aq[1], false, b1, (short)0, z, false, false);
      s[st] = z;
    }

    // Online softmax: each row lives in one 16-lane DPP row
#pragma unroll
    for (int r = 0; r < 8; ++r) {
      int qi = qbase + r + hi * 8;
      float sv[4];
#pragma unroll
      for (int st = 0; st < 4; ++st) {
        float x = s[st][r];
        if (t0 + st * 16 + ln15 > qi) x = -3.0e38f;
        sv[st] = x;
      }
      float mx = rowmax16(fmaxf(fmaxf(sv[0], sv[1]), fmaxf(sv[2], sv[3])));
      float mnew = fmaxf(mrow[r], mx);
      float al = __expf(mrow[r] - mnew);
      float p0 = __expf(sv[0] - mnew);
      float p1 = __expf(sv[1] - mnew);
      float p2 = __expf(sv[2] - mnew);
      float p3 = __expf(sv[3] - mnew);
      float rs = rowsum16((p0 + p1) + (p2 + p3));
      lrow[r] = lrow[r] * al + rs;
      mrow[r] = mnew;
#pragma unroll
      for (int dn = 0; dn < 4; ++dn) oacc[dn][r] *= al;
      int prow = r + hi * 8;
      pb[prow * 64 + ln15]      = (h16)p0;
      pb[prow * 64 + 16 + ln15] = (h16)p1;
      pb[prow * 64 + 32 + ln15] = (h16)p2;
      pb[prow * 64 + 48 + ln15] = (h16)p3;
    }
    asm volatile("" ::: "memory");  // keep same-wave LDS store->gather order

    // O += P @ V : P 16x64 = two A-fragments; 8 WMMAs
    v16h ap[2];
#pragma unroll
    for (int kc = 0; kc < 2; ++kc)
      ap[kc] = cat8(*(const v8h*)&pb[ln15 * 64 + kc * 32 + kb8],
                    *(const v8h*)&pb[ln15 * 64 + kc * 32 + kb8 + 16]);
#pragma unroll
    for (int dn = 0; dn < 4; ++dn) {
      int dcol = dn * 16 + ln15;
#pragma unroll
      for (int kc = 0; kc < 2; ++kc) {
        v16h bv = cat8(*(const v8h*)&vtS[dcol * 64 + kc * 32 + kb8],
                       *(const v8h*)&vtS[dcol * 64 + kc * 32 + kb8 + 16]);
        oacc[dn] = __builtin_amdgcn_wmma_f32_16x16x32_f16(
            false, ap[kc], false, bv, (short)0, oacc[dn], false, false);
      }
    }
  }

  // Normalize and write attention output (f16), [B,T,H,D] flat layout
#pragma unroll
  for (int dn = 0; dn < 4; ++dn)
#pragma unroll
    for (int r = 0; r < 8; ++r) {
      int row = qbase + r + hi * 8;
      float ov = oacc[dn][r] / lrow[r];
      Om[((size_t)(b * T + row)) * C + h * D + dn * 16 + ln15] = (h16)ov;
    }
}

// ---------------------------------------------------------------------------
// Host entry
// ---------------------------------------------------------------------------
extern "C" void kernel_launch(void* const* d_in, const int* in_sizes, int n_in,
                              void* d_out, int out_size, void* d_ws, size_t ws_size,
                              hipStream_t stream) {
  (void)in_sizes; (void)n_in; (void)out_size; (void)ws_size;
  const float* x     = (const float*)d_in[0];
  const float* w_qkv = (const float*)d_in[1];
  const float* w_out = (const float*)d_in[2];
  float* out = (float*)d_out;

  constexpr int Bb = 2, T = 2048, C = 1024;
  constexpr size_t MT = (size_t)Bb * T;  // 4096 rows

  h16* ws    = (h16*)d_ws;
  h16* xb    = ws;                         // MT*C
  h16* wqkvb = xb + MT * C;                // 3C*C
  h16* woutb = wqkvb + (size_t)3 * C * C;  // C*C
  h16* qb    = woutb + (size_t)C * C;      // MT*C
  h16* kb    = qb + MT * C;                // MT*C
  h16* vb    = kb + MT * C;                // MT*C
  h16* attnb = vb + MT * C;                // MT*C

  {
    int n1 = (int)(MT * C);
    cvt_f32_f16<<<(n1 / 4 + 255) / 256, 256, 0, stream>>>(x, xb, n1);
    int n2 = 3 * C * C;
    cvt_f32_f16<<<(n2 / 4 + 255) / 256, 256, 0, stream>>>(w_qkv, wqkvb, n2);
    int n3 = C * C;
    cvt_f32_f16<<<(n3 / 4 + 255) / 256, 256, 0, stream>>>(w_out, woutb, n3);
  }

  // QKV projection: [4096,1024] x [1024,3072] -> split f16 Q/K/V
  gemm_f16<true><<<dim3(3 * C / 128, (int)MT / 128), 256, 0, stream>>>(
      xb, wqkvb, nullptr, qb, kb, vb, (int)MT, 3 * C, C);

  // Causal flash attention
  flash_attn<<<dim3(T / 64, Bb * 16), 128, 0, stream>>>(qb, kb, vb, attnb);

  // Output projection: [4096,1024] x [1024,1024] -> f32 d_out
  gemm_f16<false><<<dim3(C / 128, (int)MT / 128), 256, 0, stream>>>(
      attnb, woutb, out, nullptr, nullptr, nullptr, (int)MT, C, C);
}
